// Basic_RSNN_spike_PLIF_88845693485081
// MI455X (gfx1250) — matrix operand order repaired and checked by reading, not compiled, not run on hardware
//
#include <hip/hip_runtime.h>
#include <stdint.h>

// Problem dims
#define B_    256
#define T_    250
#define NIN   700
#define NINP  704     // padded K for ff GEMM (22 * 32)
#define NH    1024
#define NOUT  20
#define KPAD  40      // LDS tile row stride in halves: 80B => 16B-aligned segs, 16 distinct banks

typedef __attribute__((ext_vector_type(16))) __bf16 v16bf;
typedef __attribute__((ext_vector_type(8)))  float  v8f;

union FragU { v16bf v; uint32_t u[8]; };

__device__ __forceinline__ uint16_t f2bf(float f) {
  union { float f; uint32_t u; } c; c.f = f;
  uint32_t u = c.u;
  uint32_t r = u + 0x7FFFu + ((u >> 16) & 1u);   // round-to-nearest-even
  return (uint16_t)(r >> 16);
}
__device__ __forceinline__ float bf2f(uint16_t h) {
  union { uint32_t u; float f; } c; c.u = ((uint32_t)h) << 16;
  return c.f;
}

// ---- CDNA5 async global->LDS copy: one b128 (16B) per lane, tracked by ASYNCcnt.
// lds_off: LDS byte offset (low 32 bits of flat shared address).
__device__ __forceinline__ void async_copy_b128(uint32_t lds_off, const void* gaddr) {
  asm volatile("global_load_async_to_lds_b128 %0, %1, off"
               :: "v"(lds_off), "v"(gaddr)
               : "memory");
}
__device__ __forceinline__ void wait_async0() {
  asm volatile("s_wait_asynccnt 0x0" ::: "memory");
}

// ---- global -> LDS tile: 64 rows x 32 halves (4 KB), 128 threads, 2 async b128 per thread
__device__ __forceinline__ void async_tile64(uint16_t* dst_lds, const uint16_t* __restrict__ src,
                                             size_t ld, size_t row0, int k0) {
  const int tid = threadIdx.x;
#pragma unroll
  for (int it = 0; it < 2; ++it) {
    int L = tid + it * 128;          // 0..255 segments of 16B
    int row = L >> 2;                // 0..63
    int seg = L & 3;                 // 4 segments of 8 halves
    const void* g = (const void*)(src + (row0 + (size_t)row) * ld + (size_t)k0 + seg * 8);
    uint32_t lo = (uint32_t)(uintptr_t)(dst_lds + row * KPAD + seg * 8);  // 16B-aligned (row*80+seg*16)
    async_copy_b128(lo, g);
  }
}

// ---- A fragment (16x32 bf16), ISA 7.12.2 layout:
// lanes 0-15: row=M=lane, K in {0..7} (v0..3) and {16..23} (v4..7)
// lanes 16-31: row=M=lane-16, K in {8..15} and {24..31}
__device__ __forceinline__ v16bf frag_a(const uint16_t* As, int mfrag, int lane) {
  FragU f;
  const int half = lane >> 4;
  const uint16_t* p = As + (mfrag + (lane & 15)) * KPAD;
  const int b0 = half * 8, b1 = 16 + half * 8;
#pragma unroll
  for (int j = 0; j < 4; ++j) f.u[j]     = *(const uint32_t*)(p + b0 + 2 * j);
#pragma unroll
  for (int j = 0; j < 4; ++j) f.u[4 + j] = *(const uint32_t*)(p + b1 + 2 * j);
  return f.v;
}

// ---- B fragment (32x16 bf16): lane = column N; lanes 0-15 hold K={0..7,16..23}, lanes 16-31 K={8..15,24..31}.
// Bs is stored N-major [64][KPAD] (K contiguous) so each VGPR is one u32 load.
__device__ __forceinline__ v16bf frag_b(const uint16_t* Bs, int nfrag, int lane) {
  FragU f;
  const int half = lane >> 4;
  const uint16_t* p = Bs + (nfrag + (lane & 15)) * KPAD + half * 8;
#pragma unroll
  for (int j = 0; j < 8; ++j) f.u[j] = *(const uint32_t*)(p + (j >> 2) * 16 + 2 * (j & 3));
  return f.v;
}

#define WMMA_BF16(a, b, c) \
  __builtin_amdgcn_wmma_f32_16x16x32_bf16(false, (a), false, (b), (short)0, (c), false, false)

// =============== Phase 1: I_ff = x @ W1^T  (bf16 WMMA, fp32 accumulate) ===============
__global__ void __launch_bounds__(128)
ff_gemm(const uint16_t* __restrict__ A,   // [M][K] bf16 (padded x)
        const uint16_t* __restrict__ Bt,  // [N][K] bf16 (W1 row-major == B^T)
        float* __restrict__ C, int K) {
  __shared__ uint16_t As[64 * KPAD];
  __shared__ uint16_t Bs[64 * KPAD];
  const size_t m0 = (size_t)blockIdx.y * 64;
  const int n0 = blockIdx.x * 64;
  const int lane = threadIdx.x & 31;
  const int wave = threadIdx.x >> 5;          // 4 waves: 2x2 of 32x32 blocks
  const int wm = (wave >> 1) * 32, wn = (wave & 1) * 32;

  v8f c00, c01, c10, c11;
#pragma unroll
  for (int r = 0; r < 8; ++r) { c00[r] = 0.f; c01[r] = 0.f; c10[r] = 0.f; c11[r] = 0.f; }

  for (int k0 = 0; k0 < K; k0 += 32) {
    __syncthreads();                               // previous iter's readers done
    async_tile64(As, A,  (size_t)K, m0, k0);       // memory -> LDS, no VGPR staging
    async_tile64(Bs, Bt, (size_t)K, (size_t)n0, k0);
    wait_async0();                                 // this wave's async copies done
    __syncthreads();                               // all waves' copies visible
    v16bf a0 = frag_a(As, wm,      lane);
    v16bf a1 = frag_a(As, wm + 16, lane);
    v16bf b0 = frag_b(Bs, wn,      lane);
    v16bf b1 = frag_b(Bs, wn + 16, lane);
    c00 = WMMA_BF16(a0, b0, c00);
    c01 = WMMA_BF16(a0, b1, c01);
    c10 = WMMA_BF16(a1, b0, c10);
    c11 = WMMA_BF16(a1, b1, c11);
  }
  const int half = lane >> 4, col = lane & 15;
#pragma unroll
  for (int r = 0; r < 8; ++r) {
    size_t mrow = m0 + wm + half * 8 + r;
    C[mrow * NH        + n0 + wn + col]      = c00[r];
    C[mrow * NH        + n0 + wn + 16 + col] = c01[r];
    C[(mrow + 16) * NH + n0 + wn + col]      = c10[r];
    C[(mrow + 16) * NH + n0 + wn + 16 + col] = c11[r];
  }
}

// =============== Per-step: I = I_ff[t] + spk @ Wrec^T, fused PLIF + spike ===============
__global__ void __launch_bounds__(128)
rec_step(const uint16_t* __restrict__ spk_cur,   // [B][NH] bf16 {0,1}
         const uint16_t* __restrict__ Wrecb,     // [NH][NH] bf16 row-major (== B^T)
         const float* __restrict__ Iff,          // [B*T][NH] fp32
         float* __restrict__ h_mem,              // [B][NH] fp32 (in/out)
         uint16_t* __restrict__ spk_next,        // [B][NH] bf16 out
         const float* __restrict__ w_h, const float* __restrict__ th_h, int t) {
  __shared__ uint16_t As[64 * KPAD];
  __shared__ uint16_t Bs[64 * KPAD];
  const size_t m0 = (size_t)blockIdx.y * 64;     // batch rows
  const int n0 = blockIdx.x * 64;                // hidden cols
  const int lane = threadIdx.x & 31;
  const int wave = threadIdx.x >> 5;
  const int wm = (wave >> 1) * 32, wn = (wave & 1) * 32;

  v8f c00, c01, c10, c11;
#pragma unroll
  for (int r = 0; r < 8; ++r) { c00[r] = 0.f; c01[r] = 0.f; c10[r] = 0.f; c11[r] = 0.f; }

  for (int k0 = 0; k0 < NH; k0 += 32) {
    __syncthreads();
    async_tile64(As, spk_cur, (size_t)NH, m0, k0);
    async_tile64(Bs, Wrecb,   (size_t)NH, (size_t)n0, k0);
    wait_async0();
    __syncthreads();
    v16bf a0 = frag_a(As, wm,      lane);
    v16bf a1 = frag_a(As, wm + 16, lane);
    v16bf b0 = frag_b(Bs, wn,      lane);
    v16bf b1 = frag_b(Bs, wn + 16, lane);
    c00 = WMMA_BF16(a0, b0, c00);
    c01 = WMMA_BF16(a0, b1, c01);
    c10 = WMMA_BF16(a1, b0, c10);
    c11 = WMMA_BF16(a1, b1, c11);
  }

  const float tau = 1.f / (1.f + __expf(-w_h[0]));
  const float th  = th_h[0];
  const int half = lane >> 4, col = lane & 15;

  auto epi = [&](v8f& c, int mi, int ni) {
#pragma unroll
    for (int r = 0; r < 8; ++r) {
      size_t b = m0 + wm + mi + half * 8 + r;
      size_t h = (size_t)(n0 + wn + ni + col);
      size_t idx = b * NH + h;
      float Iin = c[r] + Iff[(b * T_ + t) * NH + h];
      float sp  = bf2f(spk_cur[idx]);                      // exactly 0 or 1
      float m   = h_mem[idx] * tau * (1.f - sp) + Iin;     // PLIF decay + hard reset (to 0)
      float s   = (m - th) > 0.f ? 1.f : 0.f;              // Heaviside
      h_mem[idx]    = m;
      spk_next[idx] = s > 0.f ? (uint16_t)0x3F80 : (uint16_t)0;
    }
  };
  epi(c00, 0, 0); epi(c01, 0, 16); epi(c10, 16, 0); epi(c11, 16, 16);
}

// =============== Per-step output projection + output PLIF (tiny: 256x20) ===============
__global__ void __launch_bounds__(256)
out_step(const uint16_t* __restrict__ spk,      // [B][NH] bf16 (new spikes)
         const float* __restrict__ Wout,        // [NOUT][NH] fp32
         float* __restrict__ o_mem, float* __restrict__ o_spk_state,
         float* __restrict__ out,               // [B][T][NOUT]
         const float* __restrict__ w_o, const float* __restrict__ th_o, int t) {
  int idx = blockIdx.x * blockDim.x + threadIdx.x;
  if (idx >= B_ * NOUT) return;
  int b = idx / NOUT, o = idx % NOUT;
  const uint16_t* srow = spk + (size_t)b * NH;
  const float* wrow = Wout + (size_t)o * NH;
  float acc = 0.f;
#pragma unroll 4
  for (int h = 0; h < NH; ++h) acc += bf2f(srow[h]) * wrow[h];
  const float tau = 1.f / (1.f + __expf(-w_o[0]));
  const float th  = th_o[0];
  float m = o_mem[idx] * tau * (1.f - o_spk_state[idx]) + acc;
  float s = (m - th) > 0.f ? 1.f : 0.f;
  o_mem[idx] = m;
  o_spk_state[idx] = s;
  out[((size_t)b * T_ + t) * NOUT + o] = s;
}

// =============== Helpers: fp32 -> padded bf16, state zeroing ===============
__global__ void cvt_pad(const float* __restrict__ src, uint16_t* __restrict__ dst,
                        long long rows, int scols, int dcols) {
  long long total = rows * (long long)dcols;
  for (long long i = (long long)blockIdx.x * blockDim.x + threadIdx.x; i < total;
       i += (long long)gridDim.x * blockDim.x) {
    long long r = i / dcols;
    int cc = (int)(i - r * dcols);
    dst[i] = (cc < scols) ? f2bf(src[r * (long long)scols + cc]) : (uint16_t)0;
  }
}

__global__ void zero_u32(uint32_t* __restrict__ p, size_t n) {
  for (size_t i = blockIdx.x * (size_t)blockDim.x + threadIdx.x; i < n;
       i += (size_t)gridDim.x * blockDim.x) p[i] = 0u;
}

extern "C" void kernel_launch(void* const* d_in, const int* in_sizes, int n_in,
                              void* d_out, int out_size, void* d_ws, size_t ws_size,
                              hipStream_t stream) {
  (void)in_sizes; (void)n_in; (void)out_size; (void)ws_size;
  const float* x    = (const float*)d_in[0];   // [B,T,NIN]
  const float* W1   = (const float*)d_in[1];   // [NH,NIN]
  const float* Wrec = (const float*)d_in[2];   // [NH,NH]
  const float* Wout = (const float*)d_in[3];   // [NOUT,NH]
  const float* w_h  = (const float*)d_in[4];
  const float* th_h = (const float*)d_in[5];
  const float* w_o  = (const float*)d_in[6];
  const float* th_o = (const float*)d_in[7];
  float* out = (float*)d_out;

  // Workspace carve-up (256B aligned). Total ~= 358 MB.
  char* wsp = (char*)d_ws;
  auto carve = [&](size_t bytes) { void* p = (void*)wsp; wsp += (bytes + 255) & ~(size_t)255; return p; };
  uint16_t* xb    = (uint16_t*)carve((size_t)B_ * T_ * NINP * 2);  // 90 MB
  uint16_t* W1b   = (uint16_t*)carve((size_t)NH * NINP * 2);       // 1.4 MB
  uint16_t* Wrecb = (uint16_t*)carve((size_t)NH * NH * 2);         // 2 MB
  float*    Iff   = (float*)   carve((size_t)B_ * T_ * NH * 4);    // 262 MB
  float*    h_mem = (float*)   carve((size_t)B_ * NH * 4);
  uint16_t* spk0  = (uint16_t*)carve((size_t)B_ * NH * 2);
  uint16_t* spk1  = (uint16_t*)carve((size_t)B_ * NH * 2);
  float*    o_mem = (float*)   carve((size_t)B_ * NOUT * 4);
  float*    o_spk = (float*)   carve((size_t)B_ * NOUT * 4);

  // Convert inputs to bf16 (zero-padded K: 700 -> 704)
  cvt_pad<<<4096, 256, 0, stream>>>(x,    xb,    (long long)B_ * T_, NIN, NINP);
  cvt_pad<<<512,  256, 0, stream>>>(W1,   W1b,   (long long)NH,      NIN, NINP);
  cvt_pad<<<1024, 256, 0, stream>>>(Wrec, Wrecb, (long long)NH,      NH,  NH);

  // Zero recurrent state (h_mem .. o_spk are contiguous in the carve)
  size_t state_bytes = (size_t)(wsp - (char*)h_mem);
  zero_u32<<<1024, 256, 0, stream>>>((uint32_t*)h_mem, state_bytes / 4);

  // Phase 1: big feed-forward GEMM, I_ff = x @ W1^T   [64000 x 704] x [704 x 1024]
  dim3 ffgrid(NH / 64, (B_ * T_) / 64);
  ff_gemm<<<ffgrid, 128, 0, stream>>>(xb, W1b, Iff, NINP);

  // Phase 2: 250 sequential PLIF steps (recurrent WMMA GEMM fused with neuron update)
  dim3 rgrid(NH / 64, B_ / 64);
  for (int t = 0; t < T_; ++t) {
    const uint16_t* scur = (t & 1) ? spk1 : spk0;
    uint16_t*      snext = (t & 1) ? spk0 : spk1;
    rec_step<<<rgrid, 128, 0, stream>>>(scur, Wrecb, Iff, h_mem, snext, w_h, th_h, t);
    out_step<<<(B_ * NOUT + 255) / 256, 256, 0, stream>>>(snext, Wout, o_mem, o_spk,
                                                          out, w_o, th_o, t);
  }
}
